// ContrastiveLoss_32023276159237
// MI455X (gfx1250) — matrix operand order repaired and checked by reading, not compiled, not run on hardware
//
#include <hip/hip_runtime.h>
#include <hip/hip_bf16.h>
#include <math.h>

typedef __attribute__((ext_vector_type(16))) _Float16 v16h;
typedef __attribute__((ext_vector_type(8)))  _Float16 v8h;
typedef __attribute__((ext_vector_type(8)))  float    v8f;

#define D_DIM   256
#define TILE_M  128
#define TILE_N  128
#define BLOCK   256

// ---------------------------------------------------------------------------
// Kernel 1: L2-normalize rows of [query; pos] into f16 z (one wave per row).
// ---------------------------------------------------------------------------
__global__ __launch_bounds__(BLOCK)
void nk_normalize(const float* __restrict__ q, const float* __restrict__ p,
                  _Float16* __restrict__ zh, int B) {
    const int row  = blockIdx.x * (BLOCK / 32) + (threadIdx.x >> 5);
    const int lane = threadIdx.x & 31;
    const float* src = (row < B) ? (q + (size_t)row * D_DIM)
                                 : (p + (size_t)(row - B) * D_DIM);
    float v[8];
    float ss = 0.0f;
#pragma unroll
    for (int j = 0; j < 8; ++j) {
        v[j] = src[lane + 32 * j];
        ss += v[j] * v[j];
    }
#pragma unroll
    for (int off = 16; off >= 1; off >>= 1)
        ss += __shfl_xor(ss, off, 32);
    const float inv = 1.0f / fmaxf(sqrtf(ss), 1e-12f);
    _Float16* dst = zh + (size_t)row * D_DIM;
#pragma unroll
    for (int j = 0; j < 8; ++j)
        dst[lane + 32 * j] = (_Float16)(v[j] * inv);
}

// ---------------------------------------------------------------------------
// Kernel 2: positives pos_dot[i] = dot(z[i], z[i+B])  (one wave per i).
// ---------------------------------------------------------------------------
__global__ __launch_bounds__(BLOCK)
void nk_posdot(const _Float16* __restrict__ zh, float* __restrict__ pos_dot, int B) {
    const int i    = blockIdx.x * (BLOCK / 32) + (threadIdx.x >> 5);
    const int lane = threadIdx.x & 31;
    const _Float16* a = zh + (size_t)i * D_DIM;
    const _Float16* b = zh + (size_t)(i + B) * D_DIM;
    float s = 0.0f;
#pragma unroll
    for (int j = 0; j < 8; ++j)
        s += (float)a[lane + 32 * j] * (float)b[lane + 32 * j];
#pragma unroll
    for (int off = 16; off >= 1; off >>= 1)
        s += __shfl_xor(s, off, 32);
    if (lane == 0) pos_dot[i] = s;
}

// ---------------------------------------------------------------------------
// gfx1250 async global -> LDS copy helpers (ASYNCcnt-tracked, no VGPR staging)
// GVS addressing: mem = SGPR64 base + VGPR32 offset; VDST = LDS byte address.
// ---------------------------------------------------------------------------
__device__ __forceinline__ void async_tile_b128(const _Float16* gbase_tile,
                                                _Float16* lds_tile,
                                                int tid, int nbytes) {
    const unsigned long long gbase = (unsigned long long)gbase_tile;
    const unsigned lbase = (unsigned)(unsigned long long)lds_tile;
    // 16 bytes per async op per lane
    for (int off = tid * 16; off < nbytes; off += BLOCK * 16) {
        const unsigned l = lbase + (unsigned)off;
        const unsigned g = (unsigned)off;
        asm volatile("global_load_async_to_lds_b128 %0, %1, %2"
                     :: "v"(l), "v"(g), "s"(gbase) : "memory");
    }
}

__device__ __forceinline__ void wait_async_all() {
    asm volatile("s_wait_asynccnt 0x0" ::: "memory");
}

// ---------------------------------------------------------------------------
// Kernel 3: fused sim = z z^T -> exp(2*sim) -> masked row-sum via WMMA,
// with async-double-buffered B tiles.
// Grid: (N/TILE_M, colSplits). Each block: 8 waves, each wave owns 16 rows.
// ---------------------------------------------------------------------------
__global__ __launch_bounds__(BLOCK)
void nk_simexp(const _Float16* __restrict__ zh, float* __restrict__ rowsum,
               int N, int colsPerWG) {
    __shared__ _Float16 Abuf[TILE_M * D_DIM];        // 64 KB
    __shared__ _Float16 Bbuf[2][TILE_N * D_DIM];     // 2 x 64 KB

    const int tid   = threadIdx.x;
    const int wave  = tid >> 5;          // 0..7  -> M sub-tile
    const int lane  = tid & 31;
    const int ln    = lane & 15;         // N index in 16x16 tile / A row index
    const int g     = lane >> 4;         // K-half / M-half selector
    const int mbase = blockIdx.x * TILE_M;
    const int cstart = blockIdx.y * colsPerWG;
    const int m0    = wave * 16;
    const int nt    = colsPerWG / TILE_N;

    // Stage A tile + first B tile asynchronously (direct to LDS).
    async_tile_b128(zh + (size_t)mbase * D_DIM, Abuf, tid, TILE_M * D_DIM * 2);
    async_tile_b128(zh + (size_t)cstart * D_DIM, Bbuf[0], tid, TILE_N * D_DIM * 2);
    wait_async_all();
    __syncthreads();

    // Preload this wave's A fragments for all 8 K-chunks (K=32 each).
    // 16-bit A 16x32 layout (matching B): lanes 0-15 hold K {0..15}, lanes 16-31
    // hold K {16..31} for row M = lane%16 -> two contiguous 8xf16 slices.
    v16h afrag[8];
    {
        const _Float16* arow = &Abuf[(size_t)(m0 + ln) * D_DIM];
#pragma unroll
        for (int kk = 0; kk < 8; ++kk) {
            const v8h lo = *(const v8h*)(arow + kk * 32 + 16 * g);
            const v8h hi = *(const v8h*)(arow + kk * 32 + 16 * g + 8);
            v16h a;
#pragma unroll
            for (int j = 0; j < 8; ++j) { a[j] = lo[j]; a[8 + j] = hi[j]; }
            afrag[kk] = a;
        }
    }

    float acc[8] = {0.f, 0.f, 0.f, 0.f, 0.f, 0.f, 0.f, 0.f};

    for (int t = 0; t < nt; ++t) {
        const int cur = t & 1;
        const int cbase = cstart + t * TILE_N;

        // Prefetch next B tile into the other buffer (overlaps with compute;
        // that buffer's last readers were barrier-ordered at end of t-1).
        if (t + 1 < nt)
            async_tile_b128(zh + (size_t)(cbase + TILE_N) * D_DIM,
                            Bbuf[cur ^ 1], tid, TILE_N * D_DIM * 2);

#pragma unroll 1
        for (int ns = 0; ns < TILE_N / 16; ++ns) {
            v8f c = {};
            // 16-bit B 32x16 layout used here (same k order as A): lane = n + 16*khalf holds 16
            // contiguous K values of column n == row (cbase+n) of z.
            const _Float16* bcol =
                &Bbuf[cur][(size_t)(ns * 16 + ln) * D_DIM + g * 16];
#pragma unroll
            for (int kk = 0; kk < 8; ++kk) {
                const v16h b = *(const v16h*)(bcol + kk * 32);
                c = __builtin_amdgcn_wmma_f32_16x16x32_f16(
                        false, afrag[kk], false, b, (short)0, c, false, false);
            }
            // Epilogue: C layout lane holds (M = r + 8*g, N = ln).
            const int gcol = cbase + ns * 16 + ln;
#pragma unroll
            for (int r = 0; r < 8; ++r) {
                const int grow = mbase + m0 + 8 * g + r;
                const float e = __expf(2.0f * c[r]);    // T = 0.5
                acc[r] += (grow != gcol) ? e : 0.0f;    // mask = 1 - eye
            }
        }

        wait_async_all();   // this wave's prefetch landed
        __syncthreads();    // everyone's prefetch landed / compute done
    }

    // Reduce across the 16 lanes sharing each row (xor 8,4,2,1 stays in half).
#pragma unroll
    for (int r = 0; r < 8; ++r) {
        float s = acc[r];
#pragma unroll
        for (int off = 8; off >= 1; off >>= 1)
            s += __shfl_xor(s, off, 32);
        if (ln == 0)
            atomicAdd(&rowsum[mbase + m0 + 8 * g + r], s);
    }
}

// ---------------------------------------------------------------------------
// Kernel 4: loss = mean_i( log(denom_i) - 2*pos_i )   (single block).
// ---------------------------------------------------------------------------
__global__ __launch_bounds__(BLOCK)
void nk_loss(const float* __restrict__ rowsum, const float* __restrict__ pos_dot,
             float* __restrict__ out, int B) {
    __shared__ float red[BLOCK];
    const int tid = threadIdx.x;
    const int N = 2 * B;
    float s = 0.0f;
    for (int i = tid; i < N; i += BLOCK) {
        const float p = pos_dot[(i < B) ? i : (i - B)];
        s += logf(rowsum[i]) - 2.0f * p;
    }
    red[tid] = s;
    __syncthreads();
    for (int st = BLOCK / 2; st > 0; st >>= 1) {
        if (tid < st) red[tid] += red[tid + st];
        __syncthreads();
    }
    if (tid == 0) out[0] = red[0] / (float)N;
}

// ---------------------------------------------------------------------------
extern "C" void kernel_launch(void* const* d_in, const int* in_sizes, int n_in,
                              void* d_out, int out_size, void* d_ws, size_t ws_size,
                              hipStream_t stream) {
    const float* q = (const float*)d_in[0];
    const float* p = (const float*)d_in[1];
    // d_in[2] (neg) is unused by the reference loss.

    const int B = in_sizes[0] / D_DIM;   // 4096
    const int N = 2 * B;                 // 8192

    char* ws = (char*)d_ws;
    _Float16* zh      = (_Float16*)ws;                                   // N*D*2 bytes
    float*    pos_dot = (float*)(ws + (size_t)N * D_DIM * 2);            // B floats
    float*    rowsum  = (float*)(ws + (size_t)N * D_DIM * 2 + (size_t)B * 4); // N floats

    hipMemsetAsync(rowsum, 0, (size_t)N * sizeof(float), stream);

    nk_normalize<<<N / (BLOCK / 32), BLOCK, 0, stream>>>(q, p, zh, B);
    nk_posdot  <<<B / (BLOCK / 32), BLOCK, 0, stream>>>(zh, pos_dot, B);

    const int colSplits = 4;
    dim3 grid(N / TILE_M, colSplits);
    nk_simexp<<<grid, BLOCK, 0, stream>>>(zh, rowsum, N, N / colSplits);

    nk_loss<<<1, BLOCK, 0, stream>>>(rowsum, pos_dot, (float*)d_out, B);
}
